// LSTMStreamflowModel_87531433493232
// MI455X (gfx1250) — matrix operand order repaired and checked by reading, not compiled, not run on hardware
//
#include <hip/hip_runtime.h>

// ---------------------------------------------------------------------------
// CDNA5 / gfx1250 fused 2-layer LSTM + head.
//   grid = B/16 = 128 blocks, 512 threads (16 wave32 waves) per block.
//   Wave w owns N-tiles {g*16+w} (one 16-wide hidden slice, all 4 gates).
//   Weight B-fragments stream from L2 with ping-pong double buffering;
//   accumulators start from WMMA inline-0; bias folded into cell update;
//   gate nonlinearities use the hardware TRANS tanh (v_tanh_f32).
// ---------------------------------------------------------------------------

typedef __attribute__((ext_vector_type(16))) __bf16 bf16x16;
typedef __attribute__((ext_vector_type(8)))  float  v8f;
typedef __attribute__((ext_vector_type(4)))  unsigned int u32x4;
typedef __attribute__((ext_vector_type(8)))  int    i32x8;
typedef __attribute__((ext_vector_type(4)))  int    i32x4;

#if defined(__gfx1250__) && __has_builtin(__builtin_amdgcn_tensor_load_to_lds)
#define USE_TDM 1
#else
#define USE_TDM 0
#endif

// workspace layout (bytes)
#define WS_WIH0 0u         // 64 tiles * 1 kchunk * 1024B  = 64 KB   (packed bf16)
#define WS_WHH0 65536u     // 64 * 8 * 1024B               = 512 KB
#define WS_WIH1 589824u    // 512 KB
#define WS_WHH1 1114112u   // 512 KB
#define WS_B0   1638400u   // 1024 f32 combined bias layer0
#define WS_B1   1642496u   // 1024 f32 combined bias layer1

__device__ __forceinline__ unsigned f2bf_bits_rne(float f) {
  unsigned u = __float_as_uint(f);
  return (u + 0x7FFFu + ((u >> 16) & 1u)) >> 16;   // round-to-nearest-even
}
__device__ __forceinline__ unsigned short f2bf_bits_rz(float f) {
  return (unsigned short)(__float_as_uint(f) >> 16);  // truncate (cheap hot path)
}
// one-time weight repack: RNE quality
__device__ __forceinline__ unsigned pack2_rne(float lo, float hi) {
  return f2bf_bits_rne(lo) | (f2bf_bits_rne(hi) << 16);
}
// hot path: truncating pack, 2 VALU ops (shr + and_or)
__device__ __forceinline__ unsigned pack2_rz(float lo, float hi) {
  return (__float_as_uint(hi) & 0xFFFF0000u) | (__float_as_uint(lo) >> 16);
}
__device__ __forceinline__ float bf2f(unsigned short b) {
  return __uint_as_float(((unsigned)b) << 16);
}
// hardware TRANS tanh on CDNA5; fall back to exp form if builtin missing
__device__ __forceinline__ float ftanh(float x) {
#if __has_builtin(__builtin_amdgcn_tanhf)
  return __builtin_amdgcn_tanhf(x);
#elif __has_builtin(__builtin_amdgcn_tanh_f32)
  return __builtin_amdgcn_tanh_f32(x);
#else
  float e = __expf(-2.0f * x);
  return (1.0f - e) / (1.0f + e);
#endif
}
// exact identity: sigmoid(x) = 0.5 + 0.5*tanh(x/2)  (mul + tanh + fma)
__device__ __forceinline__ float fsig(float x) {
  return 0.5f + 0.5f * ftanh(0.5f * x);
}
__device__ __forceinline__ v8f vzero() {
  v8f z = {0.f, 0.f, 0.f, 0.f, 0.f, 0.f, 0.f, 0.f};
  return z;
}

// B-fragment (16-bit, 32x16): lane's 16 bf16 are 32 contiguous bytes.
__device__ __forceinline__ bf16x16 ld_frag32(const char* p) {
  union { uint4 q[2]; bf16x16 v; } u;
  u.q[0] = *(const uint4*)p;
  u.q[1] = *(const uint4*)(p + 16);
  return u.v;
}
// A-fragment (16-bit, 16x32) from LDS h stored row-major [M=16][K=256] bf16.
// lane<16: M=lane, K = kc*32 + {0..7, 16..23};  lane>=16: M=lane-16, K = +8.
__device__ __forceinline__ bf16x16 ld_a_lds(const char* base, int lm, int hi, int kc) {
  int off = lm * 512 + kc * 64 + hi * 16;
  union { uint4 q[2]; bf16x16 v; } u;
  u.q[0] = *(const uint4*)(base + off);
  u.q[1] = *(const uint4*)(base + off + 32);
  return u.v;
}
__device__ __forceinline__ v8f wmma_bf16(bf16x16 a, bf16x16 b, v8f c) {
  return __builtin_amdgcn_wmma_f32_16x16x32_bf16(false, a, false, b, (short)0, c,
                                                 false, false);
}
// 4 accumulating WMMAs (one per gate tile) sharing A.
__device__ __forceinline__ void mma4(v8f acc[4], bf16x16 Av, const bf16x16 B[4]) {
#pragma unroll
  for (int g = 0; g < 4; ++g) acc[g] = wmma_bf16(Av, B[g], acc[g]);
}
// first stage: acc = A*B + 0 (inline zero SRC2, no init movs)
__device__ __forceinline__ void mma4_init(v8f acc[4], bf16x16 Av, const bf16x16 B[4]) {
#pragma unroll
  for (int g = 0; g < 4; ++g) acc[g] = wmma_bf16(Av, B[g], vzero());
}
// load 4 gate fragments (tile ti = g*16+w) at byte offset `off` into a stream
__device__ __forceinline__ void loadB4(bf16x16 B[4], const char* base, int off) {
#pragma unroll
  for (int g = 0; g < 4; ++g) B[g] = ld_frag32(base + (size_t)g * 131072 + off);
}

// ---------------------------------------------------------------------------
// Pack weights [4H x K] f32 (row-major, gate-major rows) into WMMA B-layout,
// bf16, tile-linear:  tile = n_tile*(K/32)+kc, 256 dwords/tile, dword = l*8+v.
// ---------------------------------------------------------------------------
__global__ void pack_w(const float* __restrict__ W, unsigned* __restrict__ dst, int K) {
  int id = blockIdx.x * 256 + threadIdx.x;
  int Kc = K >> 5;
  int total = 64 * Kc * 256;
  if (id >= total) return;
  int v    = id & 7;
  int lane = (id >> 3) & 31;
  int tile = id >> 8;
  int kc   = tile % Kc;
  int nt   = tile / Kc;
  int g    = nt * 16 + (lane & 15);
  int k    = kc * 32 + ((lane >> 4) ? 8 : 0) + 2 * (v & 3) + 16 * (v >> 2);
  dst[id]  = pack2_rne(W[g * K + k], W[g * K + k + 1]);
}

__global__ void pack_bias(const float* __restrict__ a, const float* __restrict__ b,
                          float* __restrict__ dst, int n) {
  int i = blockIdx.x * 256 + threadIdx.x;
  if (i < n) dst[i] = a[i] + b[i];
}

// ---------------------------------------------------------------------------
// Fused 2-layer LSTM scan + static encoder + regressor head.
// ---------------------------------------------------------------------------
__global__ void __launch_bounds__(512) lstm_fused(
    const float* __restrict__ x_seq, const float* __restrict__ x_static,
    const char* __restrict__ ws, float* __restrict__ out,
    const float* __restrict__ Ws, const float* __restrict__ bs,
    const float* __restrict__ Wr1, const float* __restrict__ br1,
    const float* __restrict__ Wr2, const float* __restrict__ br2) {
  extern __shared__ char smem[];
  char* lds_wih0 = smem;             // 64 KB packed W_ih0 (B-fragments)
  char* lds_h0   = smem + 65536;     // 8 KB  h0 [16][256] bf16
  char* lds_h1   = smem + 73728;     // 8 KB  h1 [16][256] bf16

  const int tid = threadIdx.x;
  const int w   = tid >> 5;       // wave 0..15
  const int l   = tid & 31;       // lane
  const int lm  = l & 15;
  const int hi  = (l >> 4) & 1;
  const int b0  = blockIdx.x * 16;

  // zero h0/h1
  {
    uint4 z; z.x = z.y = z.z = z.w = 0;
    uint4* hz = (uint4*)(smem + 65536);
    for (int i = tid; i < 1024; i += 512) hz[i] = z;
  }

#if USE_TDM
  if (w == 0) {
    // TDM D# (CDNA5 ISA): 1-D copy of 16384 dwords.
    u32x4 g0 = {}; i32x8 g1 = {}; i32x4 g2 = {}; i32x4 g3 = {}; i32x8 g4 = {};
    unsigned long long ga = (unsigned long long)(ws + WS_WIH0);
    unsigned lda = (unsigned)(unsigned long long)smem;  // LDS aperture: addr[31:0] = offset
    const unsigned ND = 65536u / 4u;
    g0[0] = 1u;                                          // count=1 (valid)
    g0[1] = lda;                                         // lds_addr
    g0[2] = (unsigned)ga;                                // global_addr[31:0]
    g0[3] = (unsigned)((ga >> 32) & 0x01FFFFFFu) | 0x80000000u; // [56:32] | type=2
    g1[0] = (int)(2u << 16);                             // data_size = 4B units
    g1[1] = (int)((ND & 0xFFFFu) << 16);                 // tensor_dim0[15:0]
    g1[2] = (int)(((ND >> 16) & 0xFFFFu) | (1u << 16));  // td0[31:16] | tensor_dim1=1
    g1[3] = (int)((ND & 0xFFFFu) << 16);                 // tile_dim0 = ND
    g1[4] = 1;                                           // tile_dim1 = 1
    g1[5] = (int)ND;                                     // tensor_dim0_stride
    // clang-23 / therock form: 6 operands (extra int32x8 group, zero-filled)
    __builtin_amdgcn_tensor_load_to_lds(g0, g1, g2, g3, g4, 0);
#if __has_builtin(__builtin_amdgcn_s_wait_tensorcnt)
    __builtin_amdgcn_s_wait_tensorcnt(0);
#else
    asm volatile("s_wait_tensorcnt 0x0" ::: "memory");
#endif
  }
#else
  {
    const uint4* src = (const uint4*)(ws + WS_WIH0);
    uint4* dst = (uint4*)lds_wih0;
    for (int i = tid; i < 4096; i += 512) dst[i] = src[i];
  }
#endif
  __syncthreads();

  // Per-wave gate biases: wave w owns N = g*256 + w*16 + lm, g=0..3
  const float* bias0 = (const float*)(ws + WS_B0);
  const float* bias1 = (const float*)(ws + WS_B1);
  float bg0[4], bg1[4];
#pragma unroll
  for (int g = 0; g < 4; ++g) {
    int n = g * 256 + w * 16 + lm;
    bg0[g] = bias0[n];
    bg1[g] = bias1[n];
  }

  float c0[8] = {};   // fp32 cell state, element (M=r+8*hi, j=w*16+lm)
  float c1[8] = {};

  // per-wave base pointers into the packed weight streams (tile ti = g*16 + w):
  //   fragment(g, kc) at base + g*131072 + kc*1024
  const char* whh0 = ws + WS_WHH0 + (size_t)w * 8192 + l * 32;
  const char* wih1 = ws + WS_WIH1 + (size_t)w * 8192 + l * 32;
  const char* whh1 = ws + WS_WHH1 + (size_t)w * 8192 + l * 32;

  for (int t = 0; t < 365; ++t) {
    // ================= layer 0 =================
    v8f acc[4];

    // build A fragment for x_t (K=32) once
    union { unsigned u[8]; bf16x16 v; } Ax;
    {
      const float* xp = x_seq + ((size_t)(b0 + lm) * 365 + t) * 32;
      const int s = hi ? 8 : 0;
      float4 f0 = *(const float4*)(xp + s);
      float4 f1 = *(const float4*)(xp + s + 4);
      float4 f2 = *(const float4*)(xp + s + 16);
      float4 f3 = *(const float4*)(xp + s + 20);
      if (t + 1 < 365) __builtin_prefetch(xp + 32, 0, 0);  // global_prefetch next step
      Ax.u[0] = pack2_rz(f0.x, f0.y); Ax.u[1] = pack2_rz(f0.z, f0.w);
      Ax.u[2] = pack2_rz(f1.x, f1.y); Ax.u[3] = pack2_rz(f1.z, f1.w);
      Ax.u[4] = pack2_rz(f2.x, f2.y); Ax.u[5] = pack2_rz(f2.z, f2.w);
      Ax.u[6] = pack2_rz(f3.x, f3.y); Ax.u[7] = pack2_rz(f3.z, f3.w);
    }

    {
      bf16x16 Ba[4], Bb[4];   // ping-pong fragment buffers (never copied)
      // stage "x": B from LDS W_ih0; prefetch kc=0 of W_hh0 into Bb
#pragma unroll
      for (int g = 0; g < 4; ++g)
        Ba[g] = ld_frag32(lds_wih0 + (g * 16 + w) * 1024 + l * 32);
      loadB4(Bb, whh0, 0);
      mma4_init(acc, Ax.v, Ba);     // acc = Ax*Wih0 + 0
      // stages kc = 0..7 over W_hh0; even kc computes from Bb, odd from Ba
#pragma unroll
      for (int kc = 0; kc < 8; ++kc) {
        if ((kc & 1) == 0) {
          if (kc < 7) loadB4(Ba, whh0, (kc + 1) * 1024);
          mma4(acc, ld_a_lds(lds_h0, lm, hi, kc), Bb);
        } else {
          if (kc < 7) loadB4(Bb, whh0, (kc + 1) * 1024);
          mma4(acc, ld_a_lds(lds_h0, lm, hi, kc), Ba);
        }
      }
    }
    __syncthreads();  // all waves done reading h0_{t-1}
    // cell update (bias folded in): i,f,g,o for slice [w*16, w*16+16)
#pragma unroll
    for (int r = 0; r < 8; ++r) {
      float ig = fsig(acc[0][r] + bg0[0]);
      float fg = fsig(acc[1][r] + bg0[1]);
      float gg = ftanh(acc[2][r] + bg0[2]);
      float og = fsig(acc[3][r] + bg0[3]);
      float c  = fg * c0[r] + ig * gg;
      c0[r] = c;
      float h  = og * ftanh(c);
      int j = w * 16 + lm;
      int M = r + hi * 8;
      ((unsigned short*)lds_h0)[M * 256 + j] = f2bf_bits_rz(h);
    }
    __syncthreads();  // h0_t visible

    // ================= layer 1 =================
    // 16 stages: even i -> (W_ih1, A=h0_t), odd i -> (W_hh1, A=h1_{t-1}),
    // kc = i>>1; ping-pong Ba (even) / Bb (odd).
    {
      bf16x16 Ba[4], Bb[4];
      loadB4(Ba, wih1, 0);
#pragma unroll
      for (int i = 0; i < 16; ++i) {
        const char* nbase = ((i + 1) & 1) ? whh1 : wih1;
        int noff = ((i + 1) >> 1) * 1024;
        if ((i & 1) == 0) {
          if (i < 15) loadB4(Bb, nbase, noff);
          bf16x16 Av = ld_a_lds(lds_h0, lm, hi, i >> 1);
          if (i == 0) mma4_init(acc, Av, Ba);
          else        mma4(acc, Av, Ba);
        } else {
          if (i < 15) loadB4(Ba, nbase, noff);
          mma4(acc, ld_a_lds(lds_h1, lm, hi, i >> 1), Bb);
        }
      }
    }
    __syncthreads();  // all waves done reading h1_{t-1}
#pragma unroll
    for (int r = 0; r < 8; ++r) {
      float ig = fsig(acc[0][r] + bg1[0]);
      float fg = fsig(acc[1][r] + bg1[1]);
      float gg = ftanh(acc[2][r] + bg1[2]);
      float og = fsig(acc[3][r] + bg1[3]);
      float c  = fg * c1[r] + ig * gg;
      c1[r] = c;
      float h  = og * ftanh(c);
      int j = w * 16 + lm;
      int M = r + hi * 8;
      ((unsigned short*)lds_h1)[M * 256 + j] = f2bf_bits_rz(h);
    }
    __syncthreads();  // h1_t visible
  }

  // ================= head: static encoder + regressor =================
  if (tid < 16) {
    int b = b0 + tid;
    const unsigned short* hrow = (const unsigned short*)lds_h1 + tid * 256;
    float st[32];
    for (int u = 0; u < 32; ++u) {
      float a = bs[u];
      for (int s2 = 0; s2 < 27; ++s2) a += x_static[b * 27 + s2] * Ws[u * 27 + s2];
      st[u] = a > 0.f ? a : 0.f;
    }
    float y = br2[0];
    for (int v = 0; v < 32; ++v) {
      float a = br1[v];
      for (int k = 0; k < 256; ++k) a += bf2f(hrow[k]) * Wr1[v * 288 + k];
      for (int u = 0; u < 32; ++u) a += st[u] * Wr1[v * 288 + 256 + u];
      y += (a > 0.f ? a : 0.f) * Wr2[v];
    }
    out[b] = y;
  }
}

// ---------------------------------------------------------------------------
extern "C" void kernel_launch(void* const* d_in, const int* in_sizes, int n_in,
                              void* d_out, int out_size, void* d_ws, size_t ws_size,
                              hipStream_t stream) {
  (void)in_sizes; (void)n_in; (void)out_size; (void)ws_size;
  const float* x_seq    = (const float*)d_in[0];
  const float* x_static = (const float*)d_in[1];
  const float* W_ih0    = (const float*)d_in[2];
  const float* W_hh0    = (const float*)d_in[3];
  const float* b_ih0    = (const float*)d_in[4];
  const float* b_hh0    = (const float*)d_in[5];
  const float* W_ih1    = (const float*)d_in[6];
  const float* W_hh1    = (const float*)d_in[7];
  const float* b_ih1    = (const float*)d_in[8];
  const float* b_hh1    = (const float*)d_in[9];
  const float* Ws       = (const float*)d_in[10];
  const float* bs       = (const float*)d_in[11];
  const float* Wr1      = (const float*)d_in[12];
  const float* br1      = (const float*)d_in[13];
  const float* Wr2      = (const float*)d_in[14];
  const float* br2      = (const float*)d_in[15];
  char* ws = (char*)d_ws;

  // repack weights (bf16, WMMA B-layout) + fold biases, every call (deterministic)
  pack_w<<<64,  256, 0, stream>>>(W_ih0, (unsigned*)(ws + WS_WIH0), 32);
  pack_w<<<512, 256, 0, stream>>>(W_hh0, (unsigned*)(ws + WS_WHH0), 256);
  pack_w<<<512, 256, 0, stream>>>(W_ih1, (unsigned*)(ws + WS_WIH1), 256);
  pack_w<<<512, 256, 0, stream>>>(W_hh1, (unsigned*)(ws + WS_WHH1), 256);
  pack_bias<<<4, 256, 0, stream>>>(b_ih0, b_hh0, (float*)(ws + WS_B0), 1024);
  pack_bias<<<4, 256, 0, stream>>>(b_ih1, b_hh1, (float*)(ws + WS_B1), 1024);

  // fused scan: 128 blocks x 512 threads (16 waves), 80 KB dynamic LDS
  lstm_fused<<<128, 512, 81920, stream>>>(x_seq, x_static, ws, (float*)d_out,
                                          Ws, bs, Wr1, br1, Wr2, br2);
}
